// HypernymVisual_acc_35158602285671
// MI455X (gfx1250) — matrix operand order repaired and checked by reading, not compiled, not run on hardware
//
#include <hip/hip_runtime.h>

// ---------------------------------------------------------------------------
// Types for CDNA5 WMMA (wave32): bf16 A/B fragments (8 VGPRs), f32 C/D (8 VGPRs)
// ---------------------------------------------------------------------------
typedef __bf16 bf16;
typedef bf16  v16bf __attribute__((ext_vector_type(16)));
typedef float v8f   __attribute__((ext_vector_type(8)));

union Frag { v16bf v; uint4 u[2]; };

#define KSTRIDE 72   // LDS row stride in bf16 (144B): 16B-aligned, conflict-free

// ---------------------------------------------------------------------------
// NT GEMM:  Out[M,N] = A[M,K] * B[N,K]^T + bias[N]
//   A: f32 (GEMM1) or bf16 (GEMM2/hidden); B: f32 weights [out,in].
//   f32 operands are converted to bf16 in registers while staging to LDS.
// Block: 256 threads = 8 waves; tile 64(M) x 64(N); K-stage 64 (2 WMMA chunks),
// double-buffered LDS, ONE barrier per K-stage, steady-state loop fully peeled
// (no in-loop branches). Wave tile 32x16 -> 2 independent accumulator chains,
// 4 x v_wmma_f32_16x16x32_bf16 per wave per stage.
// ---------------------------------------------------------------------------
template <bool A_BF16>
__device__ __forceinline__ void stage_tile(
    const void* __restrict__ Aptr, const float* __restrict__ Bw,
    int K, int m_base, int n_base, int k0, int t_row, int t_seg,
    bf16* __restrict__ Adst, bf16* __restrict__ Bdst)
{
    bf16 stA[16], stB[16];
    if constexpr (A_BF16) {
        const bf16* sA = (const bf16*)Aptr + (size_t)(m_base + t_row) * K + k0 + t_seg;
        *(uint4*)&stA[0] = ((const uint4*)sA)[0];
        *(uint4*)&stA[8] = ((const uint4*)sA)[1];
    } else {
        const float* sA = (const float*)Aptr + (size_t)(m_base + t_row) * K + k0 + t_seg;
        float t[16];
        *(float4*)&t[0]  = ((const float4*)sA)[0];
        *(float4*)&t[4]  = ((const float4*)sA)[1];
        *(float4*)&t[8]  = ((const float4*)sA)[2];
        *(float4*)&t[12] = ((const float4*)sA)[3];
#pragma unroll
        for (int i = 0; i < 16; ++i) stA[i] = (bf16)t[i];
    }
    const float* sB = Bw + (size_t)(n_base + t_row) * K + k0 + t_seg;
    float tb[16];
    *(float4*)&tb[0]  = ((const float4*)sB)[0];
    *(float4*)&tb[4]  = ((const float4*)sB)[1];
    *(float4*)&tb[8]  = ((const float4*)sB)[2];
    *(float4*)&tb[12] = ((const float4*)sB)[3];
#pragma unroll
    for (int i = 0; i < 16; ++i) stB[i] = (bf16)tb[i];

    *(uint4*)&Adst[t_row * KSTRIDE + t_seg]     = *(uint4*)&stA[0];
    *(uint4*)&Adst[t_row * KSTRIDE + t_seg + 8] = *(uint4*)&stA[8];
    *(uint4*)&Bdst[t_row * KSTRIDE + t_seg]     = *(uint4*)&stB[0];
    *(uint4*)&Bdst[t_row * KSTRIDE + t_seg + 8] = *(uint4*)&stB[8];
}

template <bool A_BF16, bool OUT_BF16>
__global__ __launch_bounds__(256) void gemm_nt_wmma(
    const void* __restrict__ Aptr, const float* __restrict__ Bw,
    const float* __restrict__ bias, void* __restrict__ Out,
    int M, int N, int K)
{
    __shared__ __align__(16) bf16 As[2][64 * KSTRIDE];   // 2 x 9 KB
    __shared__ __align__(16) bf16 Bs[2][64 * KSTRIDE];   // 2 x 9 KB

    const int tid    = threadIdx.x;
    const int m_base = blockIdx.y * 64;
    const int n_base = blockIdx.x * 64;

    // staging: each thread moves one 16-elem segment of A and one of B
    const int t_row = tid >> 2;          // 0..63
    const int t_seg = (tid & 3) * 16;    // 0,16,32,48

    // wave tiling: 2(M) x 4(N) waves, each 32x16
    const int lane = tid & 31;
    const int wave = tid >> 5;
    const int wm   = (wave >> 2) * 32;
    const int wn   = (wave & 3) * 16;
    const int lr   = lane & 15;
    const int lh   = lane >> 4;

    v8f acc0 = {0.f, 0.f, 0.f, 0.f, 0.f, 0.f, 0.f, 0.f};
    v8f acc1 = acc0;

    // compute-only stage body (reads LDS buffer `buf`)
    auto compute_stage = [&](const bf16* __restrict__ Ab,
                             const bf16* __restrict__ Bb) {
#pragma unroll
        for (int c = 0; c < 2; ++c) {
            const int ko = c * 32;
            Frag b, a0, a1;
            const int bb = (wn + lr) * KSTRIDE + ko + lh * 8;
            b.u[0] = *(const uint4*)&Bb[bb];
            b.u[1] = *(const uint4*)&Bb[bb + 16];
            const int ab0 = (wm + lr) * KSTRIDE + ko + lh * 8;
            a0.u[0] = *(const uint4*)&Ab[ab0];
            a0.u[1] = *(const uint4*)&Ab[ab0 + 16];
            const int ab1 = (wm + 16 + lr) * KSTRIDE + ko + lh * 8;
            a1.u[0] = *(const uint4*)&Ab[ab1];
            a1.u[1] = *(const uint4*)&Ab[ab1 + 16];

            acc0 = __builtin_amdgcn_wmma_f32_16x16x32_bf16(
                       false, a0.v, false, b.v, (short)0, acc0, false, false);
            acc1 = __builtin_amdgcn_wmma_f32_16x16x32_bf16(
                       false, a1.v, false, b.v, (short)0, acc1, false, false);
        }
    };

    // prologue: stage k0=0 into buffer 0
    stage_tile<A_BF16>(Aptr, Bw, K, m_base, n_base, 0, t_row, t_seg,
                       &As[0][0], &Bs[0][0]);
    __syncthreads();

    // steady state: staging always enabled, no in-loop branches
    int buf = 0;
    for (int k0 = 0; k0 < K - 64; k0 += 64) {
        const int kn = k0 + 64;

        // 1) issue next-stage global loads (latency hidden by WMMAs)
        float4 ra[4], rb[4];
        uint4  rha[2];
        if constexpr (A_BF16) {
            const bf16* sA = (const bf16*)Aptr +
                             (size_t)(m_base + t_row) * K + kn + t_seg;
            rha[0] = ((const uint4*)sA)[0];
            rha[1] = ((const uint4*)sA)[1];
        } else {
            const float* sA = (const float*)Aptr +
                              (size_t)(m_base + t_row) * K + kn + t_seg;
            ra[0] = ((const float4*)sA)[0];
            ra[1] = ((const float4*)sA)[1];
            ra[2] = ((const float4*)sA)[2];
            ra[3] = ((const float4*)sA)[3];
        }
        const float* sB = Bw + (size_t)(n_base + t_row) * K + kn + t_seg;
        rb[0] = ((const float4*)sB)[0];
        rb[1] = ((const float4*)sB)[1];
        rb[2] = ((const float4*)sB)[2];
        rb[3] = ((const float4*)sB)[3];

        // 2) compute current stage: 2 chunks x 2 WMMAs
        compute_stage(&As[buf][0], &Bs[buf][0]);

        // 3) convert + store next stage into the other buffer
        bf16 stA[16], stB[16];
        if constexpr (A_BF16) {
            *(uint4*)&stA[0] = rha[0];
            *(uint4*)&stA[8] = rha[1];
        } else {
            const float* t = (const float*)ra;
#pragma unroll
            for (int i = 0; i < 16; ++i) stA[i] = (bf16)t[i];
        }
        const float* tb = (const float*)rb;
#pragma unroll
        for (int i = 0; i < 16; ++i) stB[i] = (bf16)tb[i];

        bf16* dA = &As[buf ^ 1][t_row * KSTRIDE + t_seg];
        bf16* dB = &Bs[buf ^ 1][t_row * KSTRIDE + t_seg];
        *(uint4*)(dA)     = *(uint4*)&stA[0];
        *(uint4*)(dA + 8) = *(uint4*)&stA[8];
        *(uint4*)(dB)     = *(uint4*)&stB[0];
        *(uint4*)(dB + 8) = *(uint4*)&stB[8];

        __syncthreads();   // single barrier per K-stage
        buf ^= 1;
    }

    // epilogue stage: compute only, no staging
    compute_stage(&As[buf][0], &Bs[buf][0]);

    // ---- bias add + store (C layout: VGPR g -> M = g + 8*lh) ----
    const int col = n_base + wn + lr;
    const float bv = bias[col];
#pragma unroll
    for (int g = 0; g < 8; ++g) {
        const int row = m_base + wm + g + 8 * lh;
        const float v0 = acc0[g] + bv;
        const float v1 = acc1[g] + bv;
        if constexpr (OUT_BF16) {
            ((bf16*)Out)[(size_t)row * N + col]        = (bf16)v0;
            ((bf16*)Out)[(size_t)(row + 16) * N + col] = (bf16)v1;
        } else {
            ((float*)Out)[(size_t)row * N + col]        = v0;
            ((float*)Out)[(size_t)(row + 16) * N + col] = v1;
        }
    }
}

// ---------------------------------------------------------------------------
// Positive scores: out[b*2049] = -||relu(p_lfs[b] - emb[b])||_2
// ---------------------------------------------------------------------------
__global__ void score_pos(const float* __restrict__ p_lfs,
                          const float* __restrict__ emb,
                          float* __restrict__ out)
{
    const int b = blockIdx.x * blockDim.x + threadIdx.x;
    if (b >= 128) return;
    float s = 0.f;
    for (int d = 0; d < 512; ++d) {
        float dl = fmaxf(p_lfs[(size_t)b * 512 + d] - emb[(size_t)b * 512 + d], 0.f);
        s = fmaf(dl, dl, s);
    }
    out[(size_t)b * 2049] = -sqrtf(s);
}

// ---------------------------------------------------------------------------
// Negative scores: out[b*2049 + 1 + n] = -||relu(n_lfs[n] - emb[b])||_2
// Block: 8 batch rows x 256 negatives; d in 32-wide LDS chunks.
// ---------------------------------------------------------------------------
__global__ __launch_bounds__(256) void score_neg(
    const float* __restrict__ n_lfs, const float* __restrict__ emb,
    float* __restrict__ out)
{
    __shared__ float embS[8][512];   // 16 KB
    __shared__ float nS[256][33];    // 33 KB (stride 33: conflict-free rows)

    const int tid = threadIdx.x;
    const int b0  = blockIdx.y * 8;
    const int n0  = blockIdx.x * 256;

    for (int i = tid; i < 8 * 512; i += 256)
        embS[i >> 9][i & 511] = emb[(size_t)(b0 + (i >> 9)) * 512 + (i & 511)];

    float s[8] = {};
    for (int d0 = 0; d0 < 512; d0 += 32) {
        __syncthreads();             // nS free to reuse (covers embS on iter 0)
        for (int i = tid; i < 256 * 32; i += 256)
            nS[i >> 5][i & 31] = n_lfs[(size_t)(n0 + (i >> 5)) * 512 + d0 + (i & 31)];
        __syncthreads();
#pragma unroll
        for (int c = 0; c < 32; ++c) {
            const float x = nS[tid][c];
#pragma unroll
            for (int b = 0; b < 8; ++b) {
                float dl = fmaxf(x - embS[b][d0 + c], 0.f);
                s[b] = fmaf(dl, dl, s[b]);
            }
        }
    }
#pragma unroll
    for (int b = 0; b < 8; ++b)
        out[(size_t)(b0 + b) * 2049 + 1 + (n0 + tid)] = -sqrtf(s[b]);
}

// ---------------------------------------------------------------------------
extern "C" void kernel_launch(void* const* d_in, const int* in_sizes, int n_in,
                              void* d_out, int out_size, void* d_ws, size_t ws_size,
                              hipStream_t stream)
{
    (void)in_sizes; (void)n_in; (void)out_size; (void)ws_size;
    const float* vfs   = (const float*)d_in[0];  // [128, 4096]
    const float* p_lfs = (const float*)d_in[1];  // [128, 512]
    const float* n_lfs = (const float*)d_in[2];  // [2048, 512]
    const float* W_h   = (const float*)d_in[3];  // [4096, 4096]
    const float* b_h   = (const float*)d_in[4];  // [4096]
    const float* W_e   = (const float*)d_in[5];  // [512, 4096]
    const float* b_e   = (const float*)d_in[6];  // [512]
    float* out = (float*)d_out;                  // [128, 2049]

    bf16*  hidden = (bf16*)d_ws;                                   // 1 MB
    float* emb    = (float*)((char*)d_ws + (size_t)128 * 4096 * sizeof(bf16));

    // GEMM1: hidden[128,4096] = vfs @ W_h^T + b_h   (f32 in, bf16 out)
    gemm_nt_wmma<false, true><<<dim3(4096 / 64, 128 / 64), dim3(256), 0, stream>>>(
        vfs, W_h, b_h, hidden, 128, 4096, 4096);

    // GEMM2: emb[128,512] = hidden @ W_e^T + b_e    (bf16 in, f32 out)
    gemm_nt_wmma<true, false><<<dim3(512 / 64, 128 / 64), dim3(256), 0, stream>>>(
        hidden, W_e, b_e, emb, 128, 512, 4096);

    score_pos<<<1, 128, 0, stream>>>(p_lfs, emb, out);
    score_neg<<<dim3(2048 / 256, 128 / 8), dim3(256), 0, stream>>>(n_lfs, emb, out);
}
